// WaveletPyramidCrossAttention_61452392071591
// MI455X (gfx1250) — matrix operand order repaired and checked by reading, not compiled, not run on hardware
//
#include <hip/hip_runtime.h>
#include <cstdint>
#include <cstddef>

typedef unsigned short u16;
typedef __attribute__((ext_vector_type(16))) __bf16 v16bf;
typedef __attribute__((ext_vector_type(8)))  float  v8f;
typedef unsigned int v4u __attribute__((ext_vector_type(4)));
typedef int          v4i __attribute__((ext_vector_type(4)));
typedef int          v8i __attribute__((ext_vector_type(8)));

union ABf { v16bf v; uint4 d4[2]; };

__device__ __forceinline__ u16 f2bf(float f) {
  unsigned int u = __float_as_uint(f);
  u += 0x7FFFu + ((u >> 16) & 1u);   // round-to-nearest-even
  return (u16)(u >> 16);
}

// ---------------------------------------------------------------------------
// Tensor Data Mover: 2D tile load global -> LDS (bf16 elements).
// D# layout per CDNA5 ISA ch.8: group0 = {flags, lds_addr, global_addr lo/hi+type},
// group1 = {data_size, tensor dims, tile dims, dim0 stride}.
// ---------------------------------------------------------------------------
#if defined(__has_builtin)
#if __has_builtin(__builtin_amdgcn_tensor_load_to_lds)
#define HAVE_TDM 1
#endif
#endif

#ifdef HAVE_TDM
#if __has_include(<hip/amd_detail/amd_gfx1250_TDM.h>)
// amdgpu-toolchain (therock): 6-arg form
#define TDM_CALL(g0, g1, g2, g3) \
  __builtin_amdgcn_tensor_load_to_lds((g0), (g1), (g2), (g3), (v8i){0,0,0,0,0,0,0,0}, 0)
#else
// ROCm 7.2: 5-arg form
#define TDM_CALL(g0, g1, g2, g3) \
  __builtin_amdgcn_tensor_load_to_lds((g0), (g1), (g2), (g3), 0)
#endif

__device__ __forceinline__ void tdm_load_2d(unsigned lds_addr, const void* gptr,
                                            unsigned dim0, unsigned dim1,
                                            unsigned tile0, unsigned tile1,
                                            unsigned stride0) {
  unsigned long long ga = (unsigned long long)(uintptr_t)gptr;
  v4u g0 = { 0x1u,                                   // count=1, user mode, no gather
             lds_addr,                               // lds_addr [63:32]
             (unsigned)ga,                           // global_addr [95:64]
             (unsigned)((ga >> 32) & 0x01FFFFFFu) | 0x80000000u }; // ga[56:32] | type=2
  v8i g1 = { (int)(1u << 16),                        // data_size=1 (2 bytes)
             (int)((dim0 & 0xFFFFu) << 16),          // tensor_dim0[15:0] @ bits63:48
             (int)(((dim0 >> 16) & 0xFFFFu) | ((dim1 & 0xFFFFu) << 16)),
             (int)(((dim1 >> 16) & 0xFFFFu) | ((tile0 & 0xFFFFu) << 16)),
             (int)(tile1 & 0xFFFFu),                 // tile_dim1; tile_dim2=0
             (int)stride0,                           // tensor_dim0_stride[31:0]
             0, 0 };
  v4i z4 = {0, 0, 0, 0};
  TDM_CALL(g0, g1, z4, z4);
}
#endif // HAVE_TDM

// ---------------------------------------------------------------------------
// Haar DWT, one level. Output LL (f32 or bf16) + optional bands (lh,hl,hh).
// REMAP folds the (B,N)->(N,B) transpose of key_value_multi into the read.
// ---------------------------------------------------------------------------
template <bool BANDS, bool BF16OUT, bool REMAP>
__global__ void dwt_level(const float* __restrict__ in, void* __restrict__ ll_out,
                          float* __restrict__ bands, int planes, int h2, int w2) {
  int idx = blockIdx.x * blockDim.x + threadIdx.x;
  int hw2 = h2 * w2;
  if (idx >= planes * hw2) return;
  int p   = idx / hw2;
  int rem = idx - p * hw2;
  int i = rem / w2, j = rem - i * w2;
  size_t ip = (size_t)p;
  if (REMAP) { // p=(n*B+b)*C+c -> src plane (b*N+n)*C+c  (B=2,N=4,C=256)
    int f = p >> 8, c = p & 255;
    int n = f >> 1, b = f & 1;
    ip = (size_t)((b * 4 + n) * 256 + c);
  }
  int w = 2 * w2;
  const float* s = in + ip * (size_t)(4 * hw2) + (size_t)(2 * i) * w + 2 * j;
  float x00 = s[0], x01 = s[1], x10 = s[w], x11 = s[w + 1];
  float ll = (x00 + x01 + x10 + x11) * 0.5f;
  if (BF16OUT) ((u16*)ll_out)[idx] = f2bf(ll);
  else         ((float*)ll_out)[idx] = ll;
  if (BANDS) {
    float lh = (x00 + x01 - x10 - x11) * 0.5f;
    float hl = (x00 - x01 + x10 - x11) * 0.5f;
    float hh = (x00 - x01 - x10 + x11) * 0.5f;
    size_t pb = (size_t)p * 3 * hw2 + rem;
    bands[pb]            = lh;
    bands[pb + hw2]      = hl;
    bands[pb + 2 * hw2]  = hh;
  }
}

// inverse Haar, one level
__global__ void idwt_level(const float* __restrict__ ll, const float* __restrict__ bands,
                           float* __restrict__ out, int planes, int h2, int w2) {
  int idx = blockIdx.x * blockDim.x + threadIdx.x;
  int hw2 = h2 * w2;
  if (idx >= planes * hw2) return;
  int p   = idx / hw2;
  int rem = idx - p * hw2;
  int i = rem / w2, j = rem - i * w2;
  size_t pb = (size_t)p * 3 * hw2 + rem;
  float a  = ll[idx];
  float lh = bands[pb], hl = bands[pb + hw2], hh = bands[pb + 2 * hw2];
  int w = 2 * w2;
  float* o = out + (size_t)p * 4 * hw2 + (size_t)(2 * i) * w + 2 * j;
  o[0]     = (a + lh + hl + hh) * 0.5f;
  o[1]     = (a + lh - hl - hh) * 0.5f;
  o[w]     = (a - lh + hl - hh) * 0.5f;
  o[w + 1] = (a - lh - hl + hh) * 0.5f;
}

// convert the four 256x256 weight matrices to bf16
__global__ void cvt_w4(const float* __restrict__ a, const float* __restrict__ b,
                       const float* __restrict__ c, const float* __restrict__ d,
                       u16* __restrict__ out) {
  int idx = blockIdx.x * blockDim.x + threadIdx.x;
  if (idx >= 4 * 65536) return;
  int which = idx >> 16, r = idx & 65535;
  const float* src = (which == 0) ? a : (which == 1) ? b : (which == 2) ? c : d;
  out[idx] = f2bf(src[r]);
}

// ---------------------------------------------------------------------------
// 1x1 conv as bf16 WMMA GEMM:  Out[s,o,t] = sum_c W[o,c] * X[s,c,t] + bias[o]
//   MODE 0: store bf16 head-transposed (s,h,t,d), scaled  (for Q,K)
//   MODE 1: store bf16 natural (s,c,t)                    (for V)
//   MODE 2: store f32  natural (s,c,t)                    (for Wp output)
// Grid (T/64, O/32, S), 256 threads = 8 waves (2 o-tiles x 4 t-tiles).
// ---------------------------------------------------------------------------
template <int MODE>
__global__ __launch_bounds__(256) void conv1x1_wmma(
    const u16* __restrict__ X, int x_mod, const u16* __restrict__ Wt,
    const float* __restrict__ bias, void* __restrict__ out, float scale) {
  const int T = 1024, CK = 256;
  int s = blockIdx.z;
  const u16* Xs = X + (size_t)(s % x_mod) * CK * T;
  int o_base = blockIdx.y * 32;
  int t_base = blockIdx.x * 64;
  int tid = threadIdx.x, lane = tid & 31;
  int wave = tid >> 5, wave_o = wave >> 2, wave_t = wave & 3;
  int half = lane >> 4, ln = lane & 15;

  __shared__ __align__(16) u16 lA[32 * 32];   // W tile   [o][c]
  __shared__ __align__(16) u16 lX[64 * 32];   // X tile^T [t][c]

  v8f acc = {0.f, 0.f, 0.f, 0.f, 0.f, 0.f, 0.f, 0.f};

  for (int kk = 0; kk < CK; kk += 32) {
    { // stage W tile (32x32): each thread copies 4 bf16
      int row = tid >> 3, c4 = (tid & 7) * 4;
      *(uint2*)&lA[row * 32 + c4] =
          *(const uint2*)(Wt + (size_t)(o_base + row) * CK + kk + c4);
    }
    { // stage X tile transposed: thread handles c-pair (2cp,2cp+1) x 4 t values
      int cp = tid >> 4, t4 = (tid & 15) * 4;
      uint2 va = *(const uint2*)(Xs + (size_t)(kk + 2 * cp) * T + t_base + t4);
      uint2 vb = *(const uint2*)(Xs + (size_t)(kk + 2 * cp + 1) * T + t_base + t4);
      const u16* pa = (const u16*)&va;
      const u16* pb = (const u16*)&vb;
#pragma unroll
      for (int q = 0; q < 4; ++q) {
        unsigned int packed = (unsigned int)pa[q] | ((unsigned int)pb[q] << 16);
        *(unsigned int*)&lX[(t4 + q) * 32 + 2 * cp] = packed;
      }
    }
    __syncthreads();
    ABf a, b;
    const uint4* ar = (const uint4*)&lA[(wave_o * 16 + ln) * 32];
    a.d4[0] = ar[half];     a.d4[1] = ar[2 + half];
    const uint4* br = (const uint4*)&lX[(wave_t * 16 + ln) * 32];
    b.d4[0] = br[half]; b.d4[1] = br[2 + half];
    acc = __builtin_amdgcn_wmma_f32_16x16x32_bf16(false, a.v, false, b.v,
                                                  (short)0, acc, false, false);
    __syncthreads();
  }

  int o_run = o_base + wave_o * 16 + 8 * half;
  int t = t_base + wave_t * 16 + ln;
  float vals[8];
#pragma unroll
  for (int r = 0; r < 8; ++r) vals[r] = (acc[r] + bias[o_run + r]) * scale;

  if (MODE == 0) {
    int head = o_run >> 5, d0 = o_run & 31;
    uint4 pk; u16* pp = (u16*)&pk;
#pragma unroll
    for (int r = 0; r < 8; ++r) pp[r] = f2bf(vals[r]);
    u16* O = (u16*)out;
    *(uint4*)(O + (((size_t)(s * 8 + head) * 1024 + t) * 32 + d0)) = pk;
  } else if (MODE == 1) {
    u16* O = (u16*)out;
#pragma unroll
    for (int r = 0; r < 8; ++r)
      O[(size_t)(s * 256 + o_run + r) * 1024 + t] = f2bf(vals[r]);
  } else {
    float* O = (float*)out;
#pragma unroll
    for (int r = 0; r < 8; ++r)
      O[(size_t)s * 262144 + (size_t)(o_run + r) * 1024 + t] = vals[r];
  }
}

// ---------------------------------------------------------------------------
// Flash attention per (s,h), transposed-S formulation (S' = K.Q), with
// TDM double-buffered K/V staging: tensor_load_to_lds issued by wave 0 for
// block i+1 overlaps WMMA compute on block i; sync = s_wait_tensorcnt+barrier.
// Q,K as (s,h,t,d) bf16 (Q pre-scaled by hd^-0.5), V as (s,c,t) bf16.
// Grid (8 q-blocks, 64 sh), 256 threads; each wave owns 16 query columns.
// ---------------------------------------------------------------------------
__global__ __launch_bounds__(256) void flash_attn(
    const u16* __restrict__ qT, const u16* __restrict__ kT,
    const u16* __restrict__ vN, u16* __restrict__ attnout) {
  int sh = blockIdx.y, s = sh >> 3, h = sh & 7;
  int tid = threadIdx.x, lane = tid & 31, wave = tid >> 5;
  int half = lane >> 4, ln = lane & 15;
  int q0 = blockIdx.x * 128 + wave * 16;

  const u16* qbase = qT + (size_t)sh * 1024 * 32;
  const u16* kbase = kT + (size_t)sh * 1024 * 32;
  const u16* vbase = vN + ((size_t)s * 256 + h * 32) * 1024;

  __shared__ __align__(16) u16 Kl[2][64 * 32];   // [key][d], double buffered
  __shared__ __align__(16) u16 Vl[2][32 * 64];   // [d][key], double buffered
  __shared__ __align__(16) u16 Pl[8][16 * 64];   // per-wave P' tile [q][key]

  // Q as B-fragment (d x q), constant across key blocks
  ABf qf;
  { const uint4* qr = (const uint4*)(qbase + (size_t)(q0 + ln) * 32);
    qf.d4[0] = qr[half]; qf.d4[1] = qr[2 + half]; }

  float m_run = -INFINITY, l_run = 0.f;          // per-lane scalars (query col)
  v8f olo = {0.f,0.f,0.f,0.f,0.f,0.f,0.f,0.f};   // O'[d=0..15][q]
  v8f ohi = {0.f,0.f,0.f,0.f,0.f,0.f,0.f,0.f};   // O'[d=16..31][q]

#ifdef HAVE_TDM
  if (wave == 0) {
    tdm_load_2d((unsigned)(uintptr_t)&Kl[0][0], kbase, 32, 64, 32, 64, 32);
    tdm_load_2d((unsigned)(uintptr_t)&Vl[0][0], vbase, 64, 32, 64, 32, 1024);
  }
#endif

  for (int i = 0; i < 16; ++i) {
    int p = i & 1;
    int kb = i * 64;
#ifdef HAVE_TDM
    if (wave == 0) __builtin_amdgcn_s_wait_tensorcnt(0);
    __syncthreads();                       // buf[p] ready; prev reads of buf[1-p] done
    if (i + 1 < 16 && wave == 0) {
      tdm_load_2d((unsigned)(uintptr_t)&Kl[1 - p][0],
                  kbase + (size_t)(kb + 64) * 32, 32, 64, 32, 64, 32);
      tdm_load_2d((unsigned)(uintptr_t)&Vl[1 - p][0],
                  vbase + (kb + 64), 64, 32, 64, 32, 1024);
    }
#else
    { // manual staging fallback
      int key = tid >> 2, c8 = (tid & 3) * 8;
      *(uint4*)&Kl[p][key * 32 + c8] =
          *(const uint4*)(kbase + (size_t)(kb + key) * 32 + c8);
      int d = tid >> 3, k8 = (tid & 7) * 8;
      *(uint4*)&Vl[p][d * 64 + k8] =
          *(const uint4*)(vbase + (size_t)d * 1024 + kb + k8);
    }
    __syncthreads();
#endif

    // S' = K . Q   (four 16-key tiles)
    v8f sf[4];
#pragma unroll
    for (int kt = 0; kt < 4; ++kt) {
      ABf ka;
      const uint4* kr = (const uint4*)&Kl[p][(kt * 16 + ln) * 32];
      ka.d4[0] = kr[half]; ka.d4[1] = kr[2 + half];
      v8f z = {0.f,0.f,0.f,0.f,0.f,0.f,0.f,0.f};
      sf[kt] = __builtin_amdgcn_wmma_f32_16x16x32_bf16(false, ka.v, false, qf.v,
                                                       (short)0, z, false, false);
    }

    // online softmax; key reduction runs along VGPR index + lane-half
    float mx = -INFINITY;
#pragma unroll
    for (int kt = 0; kt < 4; ++kt)
#pragma unroll
      for (int r = 0; r < 8; ++r) mx = fmaxf(mx, sf[kt][r]);
    mx = fmaxf(mx, __shfl_xor(mx, 16));
    float mnew = fmaxf(m_run, mx);
    float corr = __expf(m_run - mnew);
    m_run = mnew;

    float rs = 0.f;
#pragma unroll
    for (int kt = 0; kt < 4; ++kt)
#pragma unroll
      for (int r = 0; r < 8; ++r) {
        float pv = __expf(sf[kt][r] - mnew);
        sf[kt][r] = pv;
        rs += pv;
      }
    rs += __shfl_xor(rs, 16);
    l_run = l_run * corr + rs;
#pragma unroll
    for (int r = 0; r < 8; ++r) { olo[r] *= corr; ohi[r] *= corr; }

    // P' -> LDS with packed b128 stores (keys contiguous along VGPR index)
    u16* pw = &Pl[wave][0];
#pragma unroll
    for (int kt = 0; kt < 4; ++kt) {
      uint4 pk; u16* pp = (u16*)&pk;
#pragma unroll
      for (int r = 0; r < 8; ++r) pp[r] = f2bf(sf[kt][r]);
      *(uint4*)&pw[ln * 64 + kt * 16 + 8 * half] = pk;
    }

    // O' += V^T . P'   (two 32-key K-steps x two 16-d tiles)
    const uint4* pb4 = (const uint4*)&pw[ln * 64];
#pragma unroll
    for (int ks = 0; ks < 2; ++ks) {
      ABf pb;
      pb.d4[0] = pb4[ks * 4 + half];
      pb.d4[1] = pb4[ks * 4 + 2 + half];
#pragma unroll
      for (int dt = 0; dt < 2; ++dt) {
        ABf va;
        const uint4* vr = (const uint4*)&Vl[p][(dt * 16 + ln) * 64];
        va.d4[0] = vr[ks * 4 + half];
        va.d4[1] = vr[ks * 4 + 2 + half];
        if (dt == 0)
          olo = __builtin_amdgcn_wmma_f32_16x16x32_bf16(false, va.v, false, pb.v,
                                                        (short)0, olo, false, false);
        else
          ohi = __builtin_amdgcn_wmma_f32_16x16x32_bf16(false, va.v, false, pb.v,
                                                        (short)0, ohi, false, false);
      }
    }
#ifndef HAVE_TDM
    __syncthreads();
#endif
  }

  // normalize, store attnout[(s*256 + h*32 + d)*1024 + q]  (lane = fixed q)
  float inv_l = 1.f / l_run;
  int q = q0 + ln;
  size_t cbase = (size_t)(s * 256 + h * 32);
#pragma unroll
  for (int r = 0; r < 8; ++r) {
    int dlo = r + 8 * half;
    attnout[(cbase + dlo) * 1024 + q]      = f2bf(olo[r] * inv_l);
    attnout[(cbase + 16 + dlo) * 1024 + q] = f2bf(ohi[r] * inv_l);
  }
}

// mean over N (reference reshapes flat s as (B,N)): out[b] = mean_n p[(b*4+n)]
__global__ void mean_n(const float* __restrict__ p, float* __restrict__ out) {
  int idx = blockIdx.x * blockDim.x + threadIdx.x;
  if (idx >= 2 * 262144) return;
  int b = idx / 262144, rem = idx - b * 262144;
  size_t base = (size_t)(b * 4) * 262144 + rem;
  out[idx] = 0.25f * (p[base] + p[base + 262144] + p[base + 2 * 262144] + p[base + 3 * 262144]);
}

// --------------------------- GroupNorm (3 stages) --------------------------
// stage 1: 512 blocks, each reduces a 16384-float chunk (8 chunks per group)
__global__ __launch_bounds__(256) void gnorm_partial(const float* __restrict__ x,
                                                     float2* __restrict__ part) {
  int blk = blockIdx.x;
  size_t base = (size_t)blk * 16384;
  int tid = threadIdx.x;
  float s = 0.f, ss = 0.f;
  for (int i = tid; i < 16384; i += 256) { float v = x[base + i]; s += v; ss += v * v; }
  __shared__ float red[512];
  red[tid] = s; red[256 + tid] = ss;
  __syncthreads();
  for (int st = 128; st > 0; st >>= 1) {
    if (tid < st) { red[tid] += red[tid + st]; red[256 + tid] += red[256 + tid + st]; }
    __syncthreads();
  }
  if (tid == 0) part[blk] = make_float2(red[0], red[256]);
}

// stage 2: one small block computes (mu, rstd) per group
__global__ void gnorm_stats(const float2* __restrict__ part, float2* __restrict__ stats) {
  int g = threadIdx.x;
  if (g >= 64) return;
  float s = 0.f, ss = 0.f;
  for (int i = 0; i < 8; ++i) { float2 p = part[g * 8 + i]; s += p.x; ss += p.y; }
  float mu = s / 131072.f;
  float var = ss / 131072.f - mu * mu;
  stats[g] = make_float2(mu, rsqrtf(var + 1e-5f));
}

// stage 3: fully parallel normalize + affine + residual
__global__ void gnorm_apply(const float* __restrict__ x, const float* __restrict__ query,
                            const float* __restrict__ gamma, const float* __restrict__ beta,
                            const float2* __restrict__ stats, float* __restrict__ out) {
  int idx = blockIdx.x * blockDim.x + threadIdx.x;
  if (idx >= 2 * 256 * 16384) return;
  int c  = (idx >> 14) & 255;
  int bg = idx / 131072;
  float2 st = stats[bg];
  out[idx] = (x[idx] - st.x) * st.y * gamma[c] + beta[c] + query[idx];
}

// ---------------------------------------------------------------------------
extern "C" void kernel_launch(void* const* d_in, const int* in_sizes, int n_in,
                              void* d_out, int out_size, void* d_ws, size_t ws_size,
                              hipStream_t stream) {
  (void)in_sizes; (void)n_in; (void)out_size; (void)ws_size;
  const float* query = (const float*)d_in[0];
  const float* kvm   = (const float*)d_in[1];
  // d_in[2] (value_multi) is unused by the reference
  const float* Wq = (const float*)d_in[3];  const float* bq = (const float*)d_in[4];
  const float* Wk = (const float*)d_in[5];  const float* bk = (const float*)d_in[6];
  const float* Wv = (const float*)d_in[7];  const float* bv = (const float*)d_in[8];
  const float* Wp = (const float*)d_in[9];  const float* bp = (const float*)d_in[10];
  const float* gamma = (const float*)d_in[11];
  const float* beta  = (const float*)d_in[12];
  float* outp = (float*)d_out;

  char* ws = (char*)d_ws;
  size_t off = 0;
  auto alloc = [&](size_t bytes) { size_t o = off; off = (off + bytes + 255) & ~(size_t)255; return o; };
  size_t o_qll1  = alloc((size_t)512 * 4096 * 4);      // q LL level1  (reused as p_out, same size)
  size_t o_qb1   = alloc((size_t)512 * 3 * 4096 * 4);  // q bands level1
  size_t o_qb2   = alloc((size_t)512 * 3 * 1024 * 4);  // q bands level2
  size_t o_qylb  = alloc((size_t)512 * 1024 * 2);      // q LL level2 bf16
  size_t o_kvll1 = alloc((size_t)2048 * 4096 * 4);     // kv LL level1 (reused as out_full, same size)
  size_t o_kvylb = alloc((size_t)2048 * 1024 * 2);     // kv LL level2 bf16
  size_t o_wbf   = alloc((size_t)4 * 65536 * 2);       // Wq,Wk,Wv,Wp bf16
  size_t o_qT    = alloc((size_t)2097152 * 2);         // Q (s,h,t,d) bf16, pre-scaled
  size_t o_kT    = alloc((size_t)2097152 * 2);         // K (s,h,t,d) bf16
  size_t o_vN    = alloc((size_t)2097152 * 2);         // V (s,c,t)  bf16
  size_t o_attn  = alloc((size_t)2097152 * 2);         // attention out bf16 (s,c,t)
  size_t o_omean = alloc((size_t)524288 * 4);          // mean over N (B,C,32,32)
  size_t o_ol1   = alloc((size_t)512 * 4096 * 4);      // idwt level-1 intermediate
  size_t o_gnp   = alloc((size_t)512 * 8);             // groupnorm partials
  size_t o_gns   = alloc((size_t)64 * 8);              // groupnorm stats

  float* q_ll1  = (float*)(ws + o_qll1);
  float* q_b1   = (float*)(ws + o_qb1);
  float* q_b2   = (float*)(ws + o_qb2);
  u16*   q_ylb  = (u16*)(ws + o_qylb);
  float* kv_ll1 = (float*)(ws + o_kvll1);
  u16*   kv_ylb = (u16*)(ws + o_kvylb);
  u16*   wbf    = (u16*)(ws + o_wbf);
  u16*   qT     = (u16*)(ws + o_qT);
  u16*   kT     = (u16*)(ws + o_kT);
  u16*   vN     = (u16*)(ws + o_vN);
  u16*   attn   = (u16*)(ws + o_attn);
  float* p_out  = q_ll1;                 // reuse (q_ll1 dead after level-2 DWT)
  float* o_mean = (float*)(ws + o_omean);
  float* out_l1 = (float*)(ws + o_ol1);
  float* out_full = kv_ll1;              // reuse (kv_ll1 dead after kv level-2 DWT)
  float2* gnp   = (float2*)(ws + o_gnp);
  float2* gns   = (float2*)(ws + o_gns);

  // DWT pyramid
  dwt_level<true,  false, false><<<8192,  256, 0, stream>>>(query,  (void*)q_ll1,  q_b1, 512, 64, 64);
  dwt_level<true,  true,  false><<<2048,  256, 0, stream>>>(q_ll1,  (void*)q_ylb,  q_b2, 512, 32, 32);
  dwt_level<false, false, true ><<<32768, 256, 0, stream>>>(kvm,    (void*)kv_ll1, nullptr, 2048, 64, 64);
  dwt_level<false, true,  false><<<8192,  256, 0, stream>>>(kv_ll1, (void*)kv_ylb, nullptr, 2048, 32, 32);

  // weights to bf16
  cvt_w4<<<1024, 256, 0, stream>>>(Wq, Wk, Wv, Wp, wbf);

  // projections (WMMA bf16). Q pre-scaled by hd^-0.5 = 32^-0.5.
  dim3 cgrid(16, 8, 8);
  conv1x1_wmma<0><<<cgrid, 256, 0, stream>>>(q_ylb,  2, wbf,            bq, (void*)qT, 0.17677669529663687f);
  conv1x1_wmma<0><<<cgrid, 256, 0, stream>>>(kv_ylb, 8, wbf + 65536,    bk, (void*)kT, 1.0f);
  conv1x1_wmma<1><<<cgrid, 256, 0, stream>>>(kv_ylb, 8, wbf + 131072,   bv, (void*)vN, 1.0f);

  // flash attention (transposed-S, TDM-staged)
  flash_attn<<<dim3(8, 64), 256, 0, stream>>>(qT, kT, vN, attn);

  // Wp projection (f32 out), mean over N
  conv1x1_wmma<2><<<cgrid, 256, 0, stream>>>(attn, 8, wbf + 196608, bp, (void*)p_out, 1.0f);
  mean_n<<<2048, 256, 0, stream>>>(p_out, o_mean);

  // inverse DWT with query's bands
  idwt_level<<<2048, 256, 0, stream>>>(o_mean, q_b2, out_l1,   512, 32, 32);
  idwt_level<<<8192, 256, 0, stream>>>(out_l1, q_b1, out_full, 512, 64, 64);

  // GroupNorm + residual (3-stage, bandwidth-parallel)
  gnorm_partial<<<512, 256, 0, stream>>>(out_full, gnp);
  gnorm_stats<<<1, 64, 0, stream>>>(gnp, gns);
  gnorm_apply<<<32768, 256, 0, stream>>>(out_full, query, gamma, beta, gns, outp);
}
